// MoELayer_16518444220767
// MI455X (gfx1250) — compile-verified, matrix operand order checked by reading, not run on hardware
//
#include <hip/hip_runtime.h>
#include <hip/hip_bf16.h>

#define NE 8
#define TOPK 2
#define BB 4
#define TT 2048
#define DD 1024
#define FF 2816
#define NTOK (BB*TT)        /* 8192  */
#define NSLOT (NTOK*TOPK)   /* 16384 */

typedef __attribute__((ext_vector_type(16))) __bf16 v16bf;
typedef __attribute__((ext_vector_type(8)))  __bf16 v8bf;
typedef __attribute__((ext_vector_type(8)))  float  v8f;

__device__ __forceinline__ __bf16 f2bf(float f) {
    return (__bf16)f;                   // native v_cvt f32->bf16, RNE
}
__device__ __forceinline__ float fsilu(float x) {
    // sigmoid via single v_rcp_f32 instead of IEEE div sequence
    return x * __builtin_amdgcn_rcpf(1.0f + __expf(-x));
}
// K-group permutation so a lane's fragment is one contiguous 32B run:
// row layout [g0 g2 g1 g3]; lane half 0 reads K{0..7,16..23},
// lane half 1 reads K{8..15,24..31} -- the ISA A/B VGPR order.
__device__ __forceinline__ int gperm(int g) { return ((g & 1) << 1) | (g >> 1); }

// ---------------------------------------------------------------- utilities
__global__ void moe_zero_f32(float* p, int n) {
    int i = blockIdx.x * blockDim.x + threadIdx.x;
    if (i < n) p[i] = 0.0f;
}
__global__ void moe_zero_i32(int* p, int n) {
    int i = blockIdx.x * blockDim.x + threadIdx.x;
    if (i < n) p[i] = 0;
}
__global__ void moe_cvt_bf16(const float* __restrict__ x, __bf16* __restrict__ xb, int n) {
    int i = blockIdx.x * blockDim.x + threadIdx.x;
    if (i < n) xb[i] = f2bf(x[i]);
}

// ---------------------------------------------------------------- router
__global__ void moe_router(const float* __restrict__ x, const float* __restrict__ rw,
                           int* __restrict__ counts, int* __restrict__ tok_list,
                           float* __restrict__ w_list) {
    int t = blockIdx.x * blockDim.x + threadIdx.x;
    if (t >= NTOK) return;
    float s[NE];
#pragma unroll
    for (int e = 0; e < NE; ++e) s[e] = 0.0f;
    const float* xr = x + (size_t)t * DD;
    for (int d = 0; d < DD; ++d) {
        float xv = xr[d];
#pragma unroll
        for (int e = 0; e < NE; ++e) s[e] = __fmaf_rn(xv, rw[d * NE + e], s[e]);
    }
    float m = s[0];
#pragma unroll
    for (int e = 1; e < NE; ++e) m = fmaxf(m, s[e]);
    float p[NE], sum = 0.0f;
#pragma unroll
    for (int e = 0; e < NE; ++e) { p[e] = __expf(s[e] - m); sum += p[e]; }
    float inv = 1.0f / sum;
#pragma unroll
    for (int e = 0; e < NE; ++e) p[e] *= inv;
    int i0 = 0;
#pragma unroll
    for (int e = 1; e < NE; ++e) if (p[e] > p[i0]) i0 = e;
    int i1 = (i0 == 0) ? 1 : 0;
#pragma unroll
    for (int e = 0; e < NE; ++e) if (e != i0 && p[e] > p[i1]) i1 = e;
    float rnorm = 1.0f / (p[i0] + p[i1]);
    float w0 = p[i0] * rnorm, w1 = p[i1] * rnorm;
    int pos0 = atomicAdd(&counts[i0], 1);
    tok_list[i0 * NTOK + pos0] = t; w_list[i0 * NTOK + pos0] = w0;
    int pos1 = atomicAdd(&counts[i1], 1);
    tok_list[i1 * NTOK + pos1] = t; w_list[i1 * NTOK + pos1] = w1;
}

__global__ void moe_scan(const int* __restrict__ counts, int* __restrict__ offsets) {
    if (threadIdx.x == 0 && blockIdx.x == 0) {
        int o = 0;
        for (int e = 0; e < NE; ++e) { offsets[e] = o; o += counts[e]; }
    }
}

// ---------------------------------------------------------------- gate+up GEMM
// tile: M=32, N=128, K=32/step; 8 waves = 2(M) x 4(N); each wave: two 16x16
// subtiles per matrix sharing one A fragment -> 4 WMMAs per wave per K-step.
// LDS double-buffered; one barrier per K-step.
__global__ __launch_bounds__(256) void moe_gateup(
        const __bf16* __restrict__ xb, const float* __restrict__ Wg,
        const float* __restrict__ Wu, const int* __restrict__ counts,
        const int* __restrict__ offsets, const int* __restrict__ tok_list,
        __bf16* __restrict__ act) {
    const int e = blockIdx.z;
    const int cnt = counts[e];
    const int m0 = blockIdx.y * 32;
    if (m0 >= cnt) return;
    const int f0   = blockIdx.x * 128;
    const int tid  = threadIdx.x;
    const int lane = tid & 31;
    const int wave = tid >> 5;
    const int mw   = wave >> 2;        // 0..1
    const int nw   = wave & 3;         // 0..3
    const int ml   = lane & 15;
    const int kh   = lane >> 4;        // lane half selects fragment 32B run

    __shared__ alignas(64) __bf16 As[2][32][32];    // [buf][m][perm(k)]
    __shared__ alignas(64) __bf16 Bg[2][128][32];   // [buf][n][perm(k)]
    __shared__ alignas(64) __bf16 Bu[2][128][32];
    __shared__ int toks[32];

    if (tid < 32) {
        int m = m0 + tid;
        toks[tid] = (m < cnt) ? tok_list[e * NTOK + m] : 0;
    }
    __syncthreads();

    const size_t wbase = (size_t)e * DD * FF;
    const int am = tid >> 2, ag = tid & 3;          // A-tile task (tid<128)

    v8bf  aReg;
    float gReg[2][8], uReg[2][8];

    auto load_global = [&](int k0) {
        if (tid < 128)
            aReg = *(const v8bf*)&xb[(size_t)toks[am] * DD + k0 + ag * 8];
#pragma unroll
        for (int it = 0; it < 2; ++it) {
            int task = tid + it * 256;               // 512 (n, k-group) tasks
            int n = task & 127, g = task >> 7;
#pragma unroll
            for (int r = 0; r < 8; ++r) {
                size_t gi = wbase + (size_t)(k0 + g * 8 + r) * FF + (f0 + n);
                gReg[it][r] = Wg[gi];
                uReg[it][r] = Wu[gi];
            }
        }
    };
    auto store_lds = [&](int buf) {
        if (tid < 128)
            *(v8bf*)&As[buf][am][gperm(ag) * 8] = aReg;
#pragma unroll
        for (int it = 0; it < 2; ++it) {
            int task = tid + it * 256;
            int n = task & 127, g = task >> 7;
            v8bf tg, tu;
#pragma unroll
            for (int r = 0; r < 8; ++r) { tg[r] = f2bf(gReg[it][r]); tu[r] = f2bf(uReg[it][r]); }
            *(v8bf*)&Bg[buf][n][gperm(g) * 8] = tg;
            *(v8bf*)&Bu[buf][n][gperm(g) * 8] = tu;
        }
    };

    v8f accg0 = {}, accg1 = {}, accu0 = {}, accu1 = {};

    load_global(0);
    store_lds(0);
    __syncthreads();

    for (int k0 = 0; k0 < DD; k0 += 32) {
        const int  buf  = (k0 >> 5) & 1;
        const bool more = (k0 + 32) < DD;
        if (more) load_global(k0 + 32);              // next tile's loads in flight

        v16bf a   = *(const v16bf*)&As[buf][mw * 16 + ml][kh * 16];
        v16bf bg0 = *(const v16bf*)&Bg[buf][nw * 32 + ml][kh * 16];
        v16bf bg1 = *(const v16bf*)&Bg[buf][nw * 32 + 16 + ml][kh * 16];
        v16bf bu0 = *(const v16bf*)&Bu[buf][nw * 32 + ml][kh * 16];
        v16bf bu1 = *(const v16bf*)&Bu[buf][nw * 32 + 16 + ml][kh * 16];
        accg0 = __builtin_amdgcn_wmma_f32_16x16x32_bf16(false, a, false, bg0, (short)0, accg0, false, false);
        accg1 = __builtin_amdgcn_wmma_f32_16x16x32_bf16(false, a, false, bg1, (short)0, accg1, false, false);
        accu0 = __builtin_amdgcn_wmma_f32_16x16x32_bf16(false, a, false, bu0, (short)0, accu0, false, false);
        accu1 = __builtin_amdgcn_wmma_f32_16x16x32_bf16(false, a, false, bu1, (short)0, accu1, false, false);

        if (more) store_lds(buf ^ 1);
        __syncthreads();
    }

    const int slot0 = offsets[e] + m0;
#pragma unroll
    for (int r = 0; r < 8; ++r) {        // C layout: M = r + 8*kh, N = lane&15
        int m = mw * 16 + r + (kh << 3);
        if (m0 + m < cnt) {
            size_t base = (size_t)(slot0 + m) * FF;
            int col0 = f0 + nw * 32 + ml;
            act[base + col0]      = f2bf(fsilu(accg0[r]) * accu0[r]);
            act[base + col0 + 16] = f2bf(fsilu(accg1[r]) * accu1[r]);
        }
    }
}

// ---------------------------------------------------------------- down GEMM + weighted combine
__global__ __launch_bounds__(256) void moe_down(
        const __bf16* __restrict__ act, const float* __restrict__ Wd,
        const int* __restrict__ counts, const int* __restrict__ offsets,
        const int* __restrict__ tok_list, const float* __restrict__ w_list,
        float* __restrict__ out) {
    const int e = blockIdx.z;
    const int cnt = counts[e];
    const int m0 = blockIdx.y * 32;
    if (m0 >= cnt) return;
    const int d0   = blockIdx.x * 128;
    const int tid  = threadIdx.x;
    const int lane = tid & 31;
    const int wave = tid >> 5;
    const int mw   = wave >> 2;
    const int nw   = wave & 3;
    const int ml   = lane & 15;
    const int kh   = lane >> 4;

    __shared__ alignas(64) __bf16 As[2][32][32];
    __shared__ alignas(64) __bf16 Bd[2][128][32];
    __shared__ int   toks[32];
    __shared__ float wts[32];

    if (tid < 32) {
        int m = m0 + tid;
        toks[tid] = (m < cnt) ? tok_list[e * NTOK + m] : 0;
        wts[tid]  = (m < cnt) ? w_list[e * NTOK + m] : 0.0f;
    }
    __syncthreads();

    const int slot0 = offsets[e] + m0;
    const size_t wbase = (size_t)e * FF * DD;
    const int am = tid >> 2, ag = tid & 3;
    const int arow = (m0 + am < cnt) ? (slot0 + am) : slot0;

    v8bf  aReg;
    float dReg[2][8];

    auto load_global = [&](int k0) {
        if (tid < 128)
            aReg = *(const v8bf*)&act[(size_t)arow * FF + k0 + ag * 8];
#pragma unroll
        for (int it = 0; it < 2; ++it) {
            int task = tid + it * 256;
            int n = task & 127, g = task >> 7;
#pragma unroll
            for (int r = 0; r < 8; ++r)
                dReg[it][r] = Wd[wbase + (size_t)(k0 + g * 8 + r) * DD + (d0 + n)];
        }
    };
    auto store_lds = [&](int buf) {
        if (tid < 128)
            *(v8bf*)&As[buf][am][gperm(ag) * 8] = aReg;
#pragma unroll
        for (int it = 0; it < 2; ++it) {
            int task = tid + it * 256;
            int n = task & 127, g = task >> 7;
            v8bf td;
#pragma unroll
            for (int r = 0; r < 8; ++r) td[r] = f2bf(dReg[it][r]);
            *(v8bf*)&Bd[buf][n][gperm(g) * 8] = td;
        }
    };

    v8f acc0 = {}, acc1 = {};

    load_global(0);
    store_lds(0);
    __syncthreads();

    for (int k0 = 0; k0 < FF; k0 += 32) {
        const int  buf  = (k0 >> 5) & 1;
        const bool more = (k0 + 32) < FF;
        if (more) load_global(k0 + 32);

        v16bf a  = *(const v16bf*)&As[buf][mw * 16 + ml][kh * 16];
        v16bf b0 = *(const v16bf*)&Bd[buf][nw * 32 + ml][kh * 16];
        v16bf b1 = *(const v16bf*)&Bd[buf][nw * 32 + 16 + ml][kh * 16];
        acc0 = __builtin_amdgcn_wmma_f32_16x16x32_bf16(false, a, false, b0, (short)0, acc0, false, false);
        acc1 = __builtin_amdgcn_wmma_f32_16x16x32_bf16(false, a, false, b1, (short)0, acc1, false, false);

        if (more) store_lds(buf ^ 1);
        __syncthreads();
    }

#pragma unroll
    for (int r = 0; r < 8; ++r) {
        int m = mw * 16 + r + (kh << 3);
        if (m0 + m < cnt) {
            int t = toks[m];
            float w = wts[m];
            int col0 = d0 + nw * 32 + ml;
            atomicAdd(&out[(size_t)t * DD + col0],      w * acc0[r]);
            atomicAdd(&out[(size_t)t * DD + col0 + 16], w * acc1[r]);
        }
    }
}

// ---------------------------------------------------------------- launch
extern "C" void kernel_launch(void* const* d_in, const int* in_sizes, int n_in,
                              void* d_out, int out_size, void* d_ws, size_t ws_size,
                              hipStream_t stream) {
    const float* x  = (const float*)d_in[0];   // (B,T,D)
    const float* rw = (const float*)d_in[1];   // (D,E)
    const float* Wg = (const float*)d_in[2];   // (E,D,F)
    const float* Wu = (const float*)d_in[3];   // (E,D,F)
    const float* Wd = (const float*)d_in[4];   // (E,F,D)
    float* out = (float*)d_out;                // (B,T,D)

    auto align256 = [](size_t v) { return (v + 255) & ~(size_t)255; };
    char* p = (char*)d_ws;
    __bf16* xb      = (__bf16*)p; p += align256((size_t)NTOK * DD * sizeof(__bf16));
    int*   counts   = (int*)p;    p += align256(NE * sizeof(int));
    int*   offsets  = (int*)p;    p += align256(NE * sizeof(int));
    int*   tok_list = (int*)p;    p += align256((size_t)NE * NTOK * sizeof(int));
    float* w_list   = (float*)p;  p += align256((size_t)NE * NTOK * sizeof(float));
    __bf16* actbuf  = (__bf16*)p; p += align256((size_t)NSLOT * FF * sizeof(__bf16));
    (void)ws_size; (void)in_sizes; (void)n_in; (void)out_size;

    const int nout = NTOK * DD;
    moe_zero_f32<<<(nout + 255) / 256, 256, 0, stream>>>(out, nout);
    moe_zero_i32<<<1, 32, 0, stream>>>(counts, NE);
    moe_cvt_bf16<<<(nout + 255) / 256, 256, 0, stream>>>(x, xb, nout);
    moe_router<<<NTOK / 256, 256, 0, stream>>>(x, rw, counts, tok_list, w_list);
    moe_scan<<<1, 1, 0, stream>>>(counts, offsets);

    dim3 gg(FF / 128, NTOK / 32, NE);    // (22, 256, 8); blocks past count early-exit
    moe_gateup<<<gg, 256, 0, stream>>>(xb, Wg, Wu, counts, offsets, tok_list, actbuf);

    dim3 gd(DD / 128, NTOK / 32, NE);    // (8, 256, 8)
    moe_down<<<gd, 256, 0, stream>>>(actbuf, Wd, counts, offsets, tok_list, w_list, out);
}
